// QK_69037304316356
// MI455X (gfx1250) — matrix operand order repaired and checked by reading, not compiled, not run on hardware
//
#include <hip/hip_runtime.h>
#include <hip/hip_bf16.h>

typedef __attribute__((ext_vector_type(16))) _Float16 v16h;
typedef __attribute__((ext_vector_type(8)))  _Float16 v8h;
typedef __attribute__((ext_vector_type(8)))  float    v8f;
typedef __attribute__((ext_vector_type(4)))  unsigned int u32x4;
typedef __attribute__((ext_vector_type(8)))  int      i32x8;
typedef __attribute__((ext_vector_type(4)))  int      i32x4;

// Problem constants
#define TT 4
#define BB 16
#define CC 128
#define HH 64
#define WW 64
#define TBX (TT*BB)            // 64
#define NN (HH*WW)             // 4096
#define MM ((size_t)TBX*NN)    // 262144 positions
#define EPSV 1e-5f
#define ATTN_SCALE 0.08838834764831845f  // 128^-0.5

#if __has_builtin(__builtin_amdgcn_tensor_load_to_lds) && __has_builtin(__builtin_amdgcn_s_wait_tensorcnt)
#define USE_TDM 1
#else
#define USE_TDM 0
#endif

// ---------------------------------------------------------------------------
// Tensor Data Mover: DMA a contiguous run of `nhalf` f16 elements from global
// memory into LDS at byte offset `lds_off`. 2-D descriptor, tile = (nhalf, 1).
// D# packing per CDNA5 ISA.
// This toolchain exposes the 6-arg builtin:
//   (uint32x4 g0, int32x8 g1, int32x4 g2, int32x4 g3, int32x8 pad, i32 cpol)
// ---------------------------------------------------------------------------
#if USE_TDM
__device__ __forceinline__ void tdm_load_halfs(const void* src, unsigned lds_off,
                                               unsigned nhalf) {
    unsigned long long ga = (unsigned long long)(uintptr_t)src;
    u32x4 g0 = {
        1u,                                               // count=1, user mode
        lds_off,                                          // lds_addr (bytes)
        (unsigned)(ga & 0xFFFFFFFFu),                     // global_addr[31:0]
        (unsigned)((ga >> 32) & 0x01FFFFFFu) | (2u << 30) // global_addr[56:32], type=2
    };
    i32x8 g1;
    g1[0] = (int)(1u << 16);                               // data_size=1 (2 bytes), mask=0
    g1[1] = (int)((nhalf & 0xFFFFu) << 16);                // tensor_dim0[15:0] @ bits[63:48]
    g1[2] = (int)(((nhalf >> 16) & 0xFFFFu) | (1u << 16)); // dim0[31:16], tensor_dim1=1
    g1[3] = (int)((nhalf & 0xFFFFu) << 16);                // tile_dim0 @ bits[127:112]
    g1[4] = 1;                                             // tile_dim1=1, tile_dim2=0
    g1[5] = (int)nhalf;                                    // tensor_dim0_stride[31:0]
    g1[6] = (int)((nhalf & 0xFFFFu) << 16);                // dim1_stride[15:0] (unused)
    g1[7] = 0;
    i32x4 z4 = {0, 0, 0, 0};
    i32x8 z8 = {0, 0, 0, 0, 0, 0, 0, 0};
    __builtin_amdgcn_tensor_load_to_lds(g0, g1, z4, z4, z8, 0);
}
#endif

// Stage nhalf f16 (multiple of 256) from global into LDS; all 256 threads call.
__device__ __forceinline__ void stage_weights(const _Float16* __restrict__ src,
                                              _Float16* lds_dst, unsigned nhalf,
                                              int t, int wave) {
#if USE_TDM
    (void)t;
    if (wave == 0) {
        tdm_load_halfs(src, (unsigned)(uintptr_t)lds_dst, nhalf);
        __builtin_amdgcn_s_wait_tensorcnt(0);
    }
#else
    (void)wave;
    for (unsigned i = t; i < nhalf; i += 256) lds_dst[i] = src[i];
#endif
}

// ---------------------------------------------------------------------------
// Layout change: NCHW fp32 -> NHWC f16  (p = tb*4096 + n, channels innermost)
// ---------------------------------------------------------------------------
__global__ void k_tohalf(const float* __restrict__ X, _Float16* __restrict__ Xh) {
    int c = threadIdx.x;            // 128
    size_t p = blockIdx.x;          // 262144
    size_t tb = p >> 12, n = p & 4095;
    Xh[p * CC + c] = (_Float16)X[(tb * CC + c) * (size_t)NN + n];
}

// Pack 3x3 conv weights (co, ci, kh, kw) fp32 -> (tap, co, ci) f16, ci contiguous
__global__ void k_pack_conv(const float* __restrict__ W, _Float16* __restrict__ Wp) {
    int ci = threadIdx.x, co = blockIdx.x;
    #pragma unroll
    for (int j = 0; j < 9; ++j)
        Wp[((size_t)j * CC + co) * CC + ci] = (_Float16)W[((size_t)co * CC + ci) * 9 + j];
}

// Pack dense (co, ci) fp32 -> f16
__global__ void k_pack_mat(const float* __restrict__ W, _Float16* __restrict__ Wp) {
    int ci = threadIdx.x, co = blockIdx.x;
    Wp[(size_t)co * CC + ci] = (_Float16)W[(size_t)co * CC + ci];
}

__global__ void k_zero(float* p, int n) {
    int i = blockIdx.x * 256 + threadIdx.x;
    if (i < n) p[i] = 0.f;
}

// ---------------------------------------------------------------------------
// Per-channel stats: sum and sum-of-squares over all M positions (fp32 atomics)
// ---------------------------------------------------------------------------
__global__ void k_stats(const _Float16* __restrict__ Y, float* __restrict__ sum,
                        float* __restrict__ sumsq) {
    __shared__ float ls[CC], lq[CC];
    int c = threadIdx.x & 127;
    int half = threadIdx.x >> 7;
    size_t r0 = (size_t)blockIdx.x * 64 + (size_t)half * 32;
    float s = 0.f, q = 0.f;
    #pragma unroll 4
    for (int r = 0; r < 32; ++r) {
        float v = (float)Y[(r0 + r) * CC + c];
        s += v; q += v * v;
    }
    if (half == 0) { ls[c] = s; lq[c] = q; }
    __syncthreads();
    if (half == 1) { ls[c] += s; lq[c] += q; }
    __syncthreads();
    if (threadIdx.x < CC) {
        atomicAdd(&sum[c], ls[c]);
        atomicAdd(&sumsq[c], lq[c]);
    }
}

// scale = g * rsqrt(var+eps), shift = b - mean*scale
__global__ void k_finalize(const float* __restrict__ sum, const float* __restrict__ sumsq,
                           const float* __restrict__ g, const float* __restrict__ b,
                           float* __restrict__ scale, float* __restrict__ shift, float invM) {
    int c = threadIdx.x;
    float m = sum[c] * invM;
    float v = sumsq[c] * invM - m * m;
    float sc = g[c] * rsqrtf(v + EPSV);
    scale[c] = sc;
    shift[c] = b[c] - m * sc;
}

// ---------------------------------------------------------------------------
// Implicit-GEMM 3x3 conv with WMMA f16. Per tap: TDM stages the tap's 128x128
// f16 weight slice (32 KiB) into LDS, then 8 waves run 4 WMMA each from it.
// OOB halo lanes clamp their A pointer into a zeroed guard page (branchless).
// ---------------------------------------------------------------------------
__global__ void k_conv3x3_wmma(const _Float16* __restrict__ X,    // (TB*4096) x 128 NHWC f16
                               const _Float16* __restrict__ Wp,   // (9, co, ci) f16
                               const _Float16* __restrict__ zpad, // >=32 zeroed halfs
                               _Float16* __restrict__ Y) {
    __shared__ _Float16 lds_w[CC * CC];   // 32 KiB, one tap
    int t = threadIdx.x, wave = t >> 5, lane = t & 31;
    int lm = lane & 15, hs = lane >> 4;
    int n0 = wave * 16;
    int mt = blockIdx.x;                  // 16384 pixel tiles
    int tb = mt >> 8;
    int rem = mt & 255;
    int h = rem >> 2;
    int w0 = (rem & 3) << 4;

    v8f acc = {};
    for (int tap = 0; tap < 9; ++tap) {
        int dh = tap / 3 - 1, dw = tap % 3 - 1;
        stage_weights(Wp + (size_t)tap * CC * CC, lds_w, CC * CC, t, wave);
        __syncthreads();

        int hh = h + dh;
        int ww = w0 + lm + dw;
        bool ok = ((unsigned)hh < (unsigned)HH) && ((unsigned)ww < (unsigned)WW);
        const _Float16* Arow = X + ((size_t)((tb * HH + hh) * WW + ww)) * CC;
        const _Float16* Asafe = ok ? Arow : zpad;        // pointer cndmask, no branch
        const _Float16* Brow = lds_w + (size_t)(n0 + lm) * CC;

        #pragma unroll
        for (int kb = 0; kb < 4; ++kb) {
            int cA = (kb * 32 + hs * 8) & (ok ? ~0 : 0); // OOB lanes read zpad[0..]
            v8h a0 = *(const v8h*)(Asafe + cA);
            v8h a1 = *(const v8h*)(Asafe + (cA + (ok ? 16 : 0)));
            v16h a;
            #pragma unroll
            for (int j = 0; j < 8; ++j) { a[j] = a0[j]; a[j + 8] = a1[j]; }
            int cB = kb * 32 + hs * 8;
            v8h b0 = *(const v8h*)(Brow + cB);
            v8h b1 = *(const v8h*)(Brow + cB + 16);
            v16h b;
            #pragma unroll
            for (int j = 0; j < 8; ++j) { b[j] = b0[j]; b[j + 8] = b1[j]; }
            acc = __builtin_amdgcn_wmma_f32_16x16x32_f16(
                false, a, false, b, (short)0, acc, false, false);
        }
        __syncthreads();   // everyone done with lds_w before next tap's DMA
    }
    size_t pbase = (size_t)(tb * HH + h) * WW + w0;
    #pragma unroll
    for (int r = 0; r < 8; ++r) {
        size_t m = pbase + r + 8 * hs;     // D layout: M = r + 8*(lane>>4), N = lane&15
        Y[m * CC + n0 + lm] = (_Float16)acc[r];
    }
}

// ---------------------------------------------------------------------------
// 128x128 pointwise GEMM with WMMA. Weights staged once into LDS via TDM;
// previous BN folded as per-input-channel affine (LDS tables), optional ReLU.
// Block = 256 threads (8 waves = 8 co-tiles) x 4 m-tiles -> 16 WMMA per wave.
// ---------------------------------------------------------------------------
__global__ void k_gemm_wmma(const _Float16* __restrict__ A,    // M x 128 f16 (raw)
                            const _Float16* __restrict__ Wp,   // (co, ci) f16
                            const float* __restrict__ scale,
                            const float* __restrict__ shift,
                            _Float16* __restrict__ Y,          // M x 128 f16 (raw)
                            int relu) {
    __shared__ _Float16 lds_w[CC * CC];   // 32 KiB
    __shared__ float s_sc[CC], s_sh[CC];
    int t = threadIdx.x, wave = t >> 5, lane = t & 31;
    if (t < CC) { s_sc[t] = scale[t]; s_sh[t] = shift[t]; }
    stage_weights(Wp, lds_w, CC * CC, t, wave);
    __syncthreads();

    int lm = lane & 15, hs = lane >> 4;
    int n0 = wave * 16;
    size_t mbase = (size_t)blockIdx.x * 64;
    const _Float16* Brow = lds_w + (size_t)(n0 + lm) * CC;

    #pragma unroll 1
    for (int mi = 0; mi < 4; ++mi) {
        size_t m0 = mbase + (size_t)mi * 16;
        const _Float16* Arow = A + (m0 + lm) * CC;
        __builtin_prefetch(Arow + 16 * CC, 0, 1);
        v8f acc = {};
        #pragma unroll
        for (int kb = 0; kb < 4; ++kb) {
            int cA0 = kb * 32 + hs * 8;
            int cA1 = cA0 + 16;
            v8h a0 = *(const v8h*)(Arow + cA0);
            v8h a1 = *(const v8h*)(Arow + cA1);
            v16h a;
            #pragma unroll
            for (int j = 0; j < 8; ++j) {
                float v0 = (float)a0[j] * s_sc[cA0 + j] + s_sh[cA0 + j];
                float v1 = (float)a1[j] * s_sc[cA1 + j] + s_sh[cA1 + j];
                if (relu) { v0 = fmaxf(v0, 0.f); v1 = fmaxf(v1, 0.f); }
                a[j] = (_Float16)v0;
                a[j + 8] = (_Float16)v1;
            }
            int cB = kb * 32 + hs * 8;
            v8h b0 = *(const v8h*)(Brow + cB);
            v8h b1 = *(const v8h*)(Brow + cB + 16);
            v16h b;
            #pragma unroll
            for (int j = 0; j < 8; ++j) { b[j] = b0[j]; b[j + 8] = b1[j]; }
            acc = __builtin_amdgcn_wmma_f32_16x16x32_f16(
                false, a, false, b, (short)0, acc, false, false);
        }
        #pragma unroll
        for (int r = 0; r < 8; ++r) {
            size_t m = m0 + r + 8 * hs;
            Y[m * CC + n0 + lm] = (_Float16)acc[r];
        }
    }
}

// ---------------------------------------------------------------------------
// attn = mean_c(bn(q)) * bn(k) * SCALE ; also emit normalized k for residual.
// ---------------------------------------------------------------------------
__global__ void k_attn(const _Float16* __restrict__ Q, const _Float16* __restrict__ K,
                       const float* __restrict__ sq, const float* __restrict__ tq,
                       const float* __restrict__ sk, const float* __restrict__ tk,
                       _Float16* __restrict__ attn, _Float16* __restrict__ kn) {
    __shared__ float red[CC];
    int c = threadIdx.x;
    size_t p = blockIdx.x;
    float qn  = (float)Q[p * CC + c] * sq[c] + tq[c];
    float knv = (float)K[p * CC + c] * sk[c] + tk[c];
    red[c] = qn;
    __syncthreads();
    for (int off = 64; off > 0; off >>= 1) {
        if (c < off) red[c] += red[c + off];
        __syncthreads();
    }
    float qbar = red[0] * (1.f / 128.f);
    attn[p * CC + c] = (_Float16)(qbar * knv * ATTN_SCALE);
    kn[p * CC + c]   = (_Float16)knv;
}

// Depthwise 3-tap conv along n (SAME pad within each image's 4096 positions)
__global__ void k_dwconv(const _Float16* __restrict__ A, const float* __restrict__ Wdw,
                         _Float16* __restrict__ Y) {
    int c = threadIdx.x;
    size_t p = blockIdx.x;
    int n = (int)(p & 4095);
    float acc = 0.f;
    #pragma unroll
    for (int j = -1; j <= 1; ++j) {
        int nn = n + j;
        if ((unsigned)nn < (unsigned)NN)
            acc += (float)A[(p + j) * CC + c] * Wdw[c * 3 + (j + 1)];
    }
    Y[p * CC + c] = (_Float16)acc;
}

// out = bn(Yraw) + R   (residual add with BN fold)
__global__ void k_res(const _Float16* __restrict__ Yraw, const float* __restrict__ s,
                      const float* __restrict__ tsh, const _Float16* __restrict__ R,
                      _Float16* __restrict__ Out) {
    int c = threadIdx.x;
    size_t p = blockIdx.x;
    float v = (float)Yraw[p * CC + c] * s[c] + tsh[c] + (float)R[p * CC + c];
    Out[p * CC + c] = (_Float16)v;
}

// final = bn(z2) + bn(r1), written NCHW fp32 into d_out
__global__ void k_final(const _Float16* __restrict__ Z2, const float* __restrict__ s2,
                        const float* __restrict__ t2, const _Float16* __restrict__ R1,
                        const float* __restrict__ sn, const float* __restrict__ tn,
                        float* __restrict__ Out) {
    int c = threadIdx.x;
    size_t p = blockIdx.x;
    size_t tb = p >> 12, n = p & 4095;
    float v = (float)Z2[p * CC + c] * s2[c] + t2[c]
            + ((float)R1[p * CC + c] * sn[c] + tn[c]);
    Out[(tb * CC + c) * (size_t)NN + n] = v;
}

// ---------------------------------------------------------------------------
extern "C" void kernel_launch(void* const* d_in, const int* in_sizes, int n_in,
                              void* d_out, int out_size, void* d_ws, size_t ws_size,
                              hipStream_t stream) {
    (void)in_sizes; (void)n_in; (void)out_size; (void)ws_size;
    const float* x    = (const float*)d_in[0];
    const float* wq   = (const float*)d_in[1];
    const float* gq   = (const float*)d_in[2];
    const float* bq   = (const float*)d_in[3];
    const float* wk   = (const float*)d_in[4];
    const float* gk   = (const float*)d_in[5];
    const float* bk   = (const float*)d_in[6];
    const float* w_dw = (const float*)d_in[7];
    const float* g_dw = (const float*)d_in[8];
    const float* b_dw = (const float*)d_in[9];
    const float* w_pw = (const float*)d_in[10];
    const float* g_pw = (const float*)d_in[11];
    const float* b_pw = (const float*)d_in[12];
    const float* cw1  = (const float*)d_in[13];
    const float* cg1  = (const float*)d_in[14];
    const float* cb1  = (const float*)d_in[15];
    const float* cw2  = (const float*)d_in[16];
    const float* cg2  = (const float*)d_in[17];
    const float* cb2  = (const float*)d_in[18];
    const float* mw1  = (const float*)d_in[19];
    const float* mg1  = (const float*)d_in[20];
    const float* mb1  = (const float*)d_in[21];
    const float* mw2  = (const float*)d_in[22];
    const float* mg2  = (const float*)d_in[23];
    const float* mb2  = (const float*)d_in[24];
    const float* ng   = (const float*)d_in[25];
    const float* nb   = (const float*)d_in[26];
    float* out = (float*)d_out;

    // ---- workspace: 6 aliased 64-MiB f16 activation slots + weights + stats + zpad
    const size_t ACT = MM * CC * sizeof(_Float16);     // 67,108,864 B
    char* ws = (char*)d_ws;
    _Float16* S0 = (_Float16*)(ws + 0 * ACT);   // xh -> attn
    _Float16* S1 = (_Float16*)(ws + 1 * ACT);   // qraw -> dwr -> r1
    _Float16* S2 = (_Float16*)(ws + 2 * ACT);   // kraw -> pwr -> z2
    _Float16* S3 = (_Float16*)(ws + 3 * ACT);   // kn
    _Float16* S4 = (_Float16*)(ws + 4 * ACT);   // y1 -> z1
    _Float16* S5 = (_Float16*)(ws + 5 * ACT);   // y2
    char* wr = ws + 6 * ACT;
    _Float16* wqp  = (_Float16*)wr;                   wr += 9 * CC * CC * sizeof(_Float16);
    _Float16* wkp  = (_Float16*)wr;                   wr += 9 * CC * CC * sizeof(_Float16);
    _Float16* wpwp = (_Float16*)wr;                   wr += CC * CC * sizeof(_Float16);
    _Float16* wc1p = (_Float16*)wr;                   wr += CC * CC * sizeof(_Float16);
    _Float16* wc2p = (_Float16*)wr;                   wr += CC * CC * sizeof(_Float16);
    _Float16* wm1p = (_Float16*)wr;                   wr += CC * CC * sizeof(_Float16);
    _Float16* wm2p = (_Float16*)wr;                   wr += CC * CC * sizeof(_Float16);
    float* zpadf = (float*)wr;                        wr += 128 * sizeof(float); // zero guard
    float* ST = (float*)wr;   // 9 stages x {sum,sumsq,scale,shift} x 128
    auto SUM   = [&](int s) { return ST + (size_t)s * 512; };
    auto SSQ   = [&](int s) { return ST + (size_t)s * 512 + 128; };
    auto SCALE = [&](int s) { return ST + (size_t)s * 512 + 256; };
    auto SHIFT = [&](int s) { return ST + (size_t)s * 512 + 384; };
    const _Float16* zpad = (const _Float16*)zpadf;
    // stage ids: 0=q 1=k 2=dw 3=pw 4=c1 5=c2 6=norm 7=m1 8=m2

    const float invM = 1.0f / (float)MM;
    dim3 gP((unsigned)MM), bP(CC);               // per-position elementwise kernels
    dim3 gC((unsigned)(MM / 16)), bC(256);       // conv WMMA (16-pixel tiles)
    dim3 gG((unsigned)(MM / 64)), bG(256);       // GEMM WMMA (64-row blocks)
    dim3 gS((unsigned)(MM / 64)), bS(256);       // stats

    k_zero<<<dim3(18), dim3(256), 0, stream>>>(ST, 9 * 512);
    k_zero<<<dim3(1), dim3(256), 0, stream>>>(zpadf, 128);

    // pack weights + layout conversion
    k_tohalf<<<gP, bP, 0, stream>>>(x, S0);
    k_pack_conv<<<dim3(CC), dim3(CC), 0, stream>>>(wq, wqp);
    k_pack_conv<<<dim3(CC), dim3(CC), 0, stream>>>(wk, wkp);
    k_pack_mat<<<dim3(CC), dim3(CC), 0, stream>>>(w_pw, wpwp);
    k_pack_mat<<<dim3(CC), dim3(CC), 0, stream>>>(cw1, wc1p);
    k_pack_mat<<<dim3(CC), dim3(CC), 0, stream>>>(cw2, wc2p);
    k_pack_mat<<<dim3(CC), dim3(CC), 0, stream>>>(mw1, wm1p);
    k_pack_mat<<<dim3(CC), dim3(CC), 0, stream>>>(mw2, wm2p);

    // q, k convs (implicit-GEMM WMMA, TDM-staged weights) + BN stats
    k_conv3x3_wmma<<<gC, bC, 0, stream>>>(S0, wqp, zpad, S1);
    k_conv3x3_wmma<<<gC, bC, 0, stream>>>(S0, wkp, zpad, S2);
    k_stats<<<gS, bS, 0, stream>>>(S1, SUM(0), SSQ(0));
    k_stats<<<gS, bS, 0, stream>>>(S2, SUM(1), SSQ(1));
    k_finalize<<<dim3(1), dim3(CC), 0, stream>>>(SUM(0), SSQ(0), gq, bq, SCALE(0), SHIFT(0), invM);
    k_finalize<<<dim3(1), dim3(CC), 0, stream>>>(SUM(1), SSQ(1), gk, bk, SCALE(1), SHIFT(1), invM);

    // attention fuse (also emits normalized k for the residual)
    k_attn<<<gP, bP, 0, stream>>>(S1, S2, SCALE(0), SHIFT(0), SCALE(1), SHIFT(1), S0, S3);

    // ds_conv: depthwise + BN, pointwise (WMMA) + BN
    k_dwconv<<<gP, bP, 0, stream>>>(S0, w_dw, S1);
    k_stats<<<gS, bS, 0, stream>>>(S1, SUM(2), SSQ(2));
    k_finalize<<<dim3(1), dim3(CC), 0, stream>>>(SUM(2), SSQ(2), g_dw, b_dw, SCALE(2), SHIFT(2), invM);
    k_gemm_wmma<<<gG, bG, 0, stream>>>(S1, wpwp, SCALE(2), SHIFT(2), S2, 0);
    k_stats<<<gS, bS, 0, stream>>>(S2, SUM(3), SSQ(3));
    k_finalize<<<dim3(1), dim3(CC), 0, stream>>>(SUM(3), SSQ(3), g_pw, b_pw, SCALE(3), SHIFT(3), invM);

    // MLP #1: gemm(cw1) -> BN -> ReLU -> gemm(cw2) -> BN, + k residual
    k_gemm_wmma<<<gG, bG, 0, stream>>>(S2, wc1p, SCALE(3), SHIFT(3), S4, 0);
    k_stats<<<gS, bS, 0, stream>>>(S4, SUM(4), SSQ(4));
    k_finalize<<<dim3(1), dim3(CC), 0, stream>>>(SUM(4), SSQ(4), cg1, cb1, SCALE(4), SHIFT(4), invM);
    k_gemm_wmma<<<gG, bG, 0, stream>>>(S4, wc2p, SCALE(4), SHIFT(4), S5, 1);
    k_stats<<<gS, bS, 0, stream>>>(S5, SUM(5), SSQ(5));
    k_finalize<<<dim3(1), dim3(CC), 0, stream>>>(SUM(5), SSQ(5), cg2, cb2, SCALE(5), SHIFT(5), invM);
    k_res<<<gP, bP, 0, stream>>>(S5, SCALE(5), SHIFT(5), S3, S1);   // r1 = bn(y2) + kn

    // channel norm stats on r1
    k_stats<<<gS, bS, 0, stream>>>(S1, SUM(6), SSQ(6));
    k_finalize<<<dim3(1), dim3(CC), 0, stream>>>(SUM(6), SSQ(6), ng, nb, SCALE(6), SHIFT(6), invM);

    // MLP #2 (input = normalized r1, folded), + normalized r1 residual
    k_gemm_wmma<<<gG, bG, 0, stream>>>(S1, wm1p, SCALE(6), SHIFT(6), S4, 0);
    k_stats<<<gS, bS, 0, stream>>>(S4, SUM(7), SSQ(7));
    k_finalize<<<dim3(1), dim3(CC), 0, stream>>>(SUM(7), SSQ(7), mg1, mb1, SCALE(7), SHIFT(7), invM);
    k_gemm_wmma<<<gG, bG, 0, stream>>>(S4, wm2p, SCALE(7), SHIFT(7), S2, 1);
    k_stats<<<gS, bS, 0, stream>>>(S2, SUM(8), SSQ(8));
    k_finalize<<<dim3(1), dim3(CC), 0, stream>>>(SUM(8), SSQ(8), mg2, mb2, SCALE(8), SHIFT(8), invM);

    k_final<<<gP, bP, 0, stream>>>(S2, SCALE(8), SHIFT(8), S1, SCALE(6), SHIFT(6), out);
}
